// DsFormer_35158602285125
// MI455X (gfx1250) — matrix operand order repaired and checked by reading, not compiled, not run on hardware
//
#include <hip/hip_runtime.h>
#include <math.h>

typedef __attribute__((ext_vector_type(16))) __bf16 bf16x16;
typedef __attribute__((ext_vector_type(8)))  float  floatx8;
typedef unsigned short u16;
typedef int int4v __attribute__((vector_size(16)));

#define B_    16
#define T_    512
#define FIN_  32
#define D_    512
#define H_    8
#define DH_   64
#define DFF_  2048
#define NTOP_ 35
#define PRED_ 96

#if defined(__has_builtin)
#if __has_builtin(__builtin_amdgcn_global_load_async_to_lds_b128) && \
    __has_builtin(__builtin_amdgcn_s_wait_asynccnt)
#define HAVE_ASYNC 1
#endif
#endif
#ifndef HAVE_ASYNC
#define HAVE_ASYNC 0
#endif

#define GLB_I4(p) ((int4v*)(p))
#define LDS_I4(p) ((__attribute__((address_space(3))) int4v*)(p))

// ---------------------------------------------------------------------------
// fp32 -> bf16 (round-to-nearest-even)
// ---------------------------------------------------------------------------
__device__ __forceinline__ u16 f2bf(float x) {
  unsigned int u = __float_as_uint(x);
  unsigned int r = u + 0x7FFFu + ((u >> 16) & 1u);
  return (u16)(r >> 16);
}

__global__ __launch_bounds__(256) void convert_bf16_kernel(const float* __restrict__ src,
                                                           u16* __restrict__ dst, int n) {
  int i = blockIdx.x * 256 + threadIdx.x;
  if (i < n) dst[i] = f2bf(src[i]);
}

// ---------------------------------------------------------------------------
// WMMA fragment assembly per CDNA5 16-bit layouts (ISA 7.12.2)
// A 16x32: lane L holds row L%16; lanes<16 K-base 0, lanes>=16 K-base 8 (+16 hi)
// B 32x16: lane L holds col L%16; lanes<16 K-base 0, lanes>=16 K-base 8 (+16 hi)
// ---------------------------------------------------------------------------
__device__ __forceinline__ bf16x16 fragA(const u16* As, int rowBase, int lane) {
  int r  = rowBase + (lane & 15);
  int kb = (lane < 16) ? 0 : 8;
  union { uint4 q[2]; bf16x16 v; } c;
  c.q[0] = *reinterpret_cast<const uint4*>(As + r * 32 + kb);
  c.q[1] = *reinterpret_cast<const uint4*>(As + r * 32 + kb + 16);
  return c.v;
}

__device__ __forceinline__ bf16x16 fragB(const u16* Bs, int colBase, int lane) {
  int n  = colBase + (lane & 15);
  int kb = (lane < 16) ? 0 : 8;
  union { uint4 q[2]; bf16x16 v; } c;
  c.q[0] = *reinterpret_cast<const uint4*>(Bs + n * 32 + kb);
  c.q[1] = *reinterpret_cast<const uint4*>(Bs + n * 32 + kb + 16);
  return c.v;
}

__device__ __forceinline__ floatx8 wmma_bf16(bf16x16 a, bf16x16 b, floatx8 c) {
  return __builtin_amdgcn_wmma_f32_16x16x32_bf16(false, a, false, b, (short)0, c, false, false);
}

__device__ __forceinline__ float gelu_exact(float x) {
  return 0.5f * x * (1.0f + erff(x * 0.70710678118654752f));
}

// ---------------------------------------------------------------------------
// Tiled GEMM: C[M,N] = act(A[M,K] (bf16) * W[N,K]^T (bf16) + bias[N])
// BM=128 BN=64 BK=32; 256 threads = 8 waves; wave -> 32x32 of C (2x2 WMMA).
// Double-buffered LDS; async global->LDS staging (ASYNCcnt) when available,
// else synchronous uint4 copies. K multiple of 32. M multiple of 128 at all
// call sites; A/B row indices clamped (stale columns only feed unwritten C).
// ---------------------------------------------------------------------------
template <int ACT> // 0 = none, 1 = exact GELU
__global__ __launch_bounds__(256) void gemm_wmma(const u16* __restrict__ A,
                                                 const u16* __restrict__ W,
                                                 const float* __restrict__ bias,
                                                 float* __restrict__ C,
                                                 int M, int N, int K) {
  __shared__ u16 As[2][128 * 32];
  __shared__ u16 Bs[2][64 * 32];

  const int t    = threadIdx.x;
  const int lane = t & 31;
  const int wid  = t >> 5;
  const int wm   = (wid & 3) * 32;
  const int wn   = (wid >> 2) * 32;
  const int m0   = blockIdx.y * 128;
  const int n0   = blockIdx.x * 64;

  // Staging coordinates (one thread = 32B of A tile + 16B of B tile)
  const int arow = t >> 1, akh = (t & 1) * 16;
  int gr = m0 + arow; gr = (gr < M) ? gr : (M - 1);
  const int brow = t >> 2, bkoff = (t & 3) * 8;
  int gn = n0 + brow; gn = (gn < N) ? gn : (N - 1);
  const u16* aptr = A + (size_t)gr * K + akh;
  const u16* bptr = W + (size_t)gn * K + bkoff;
  u16* asDst = &As[0][arow * 32 + akh];
  u16* bsDst = &Bs[0][brow * 32 + bkoff];
  const int abufStride = 128 * 32;
  const int bbufStride = 64 * 32;

  floatx8 acc00 = {0,0,0,0,0,0,0,0};
  floatx8 acc01 = {0,0,0,0,0,0,0,0};
  floatx8 acc10 = {0,0,0,0,0,0,0,0};
  floatx8 acc11 = {0,0,0,0,0,0,0,0};

  auto stage = [&](int p, int k0) {
#if HAVE_ASYNC
    __builtin_amdgcn_global_load_async_to_lds_b128(
        GLB_I4(aptr + k0), LDS_I4(asDst + p * abufStride), 0, 0);
    __builtin_amdgcn_global_load_async_to_lds_b128(
        GLB_I4(aptr + k0 + 8), LDS_I4(asDst + p * abufStride + 8), 0, 0);
    __builtin_amdgcn_global_load_async_to_lds_b128(
        GLB_I4(bptr + k0), LDS_I4(bsDst + p * bbufStride), 0, 0);
#else
    uint4 a0 = *reinterpret_cast<const uint4*>(aptr + k0);
    uint4 a1 = *reinterpret_cast<const uint4*>(aptr + k0 + 8);
    uint4 b0 = *reinterpret_cast<const uint4*>(bptr + k0);
    *reinterpret_cast<uint4*>(asDst + p * abufStride)     = a0;
    *reinterpret_cast<uint4*>(asDst + p * abufStride + 8) = a1;
    *reinterpret_cast<uint4*>(bsDst + p * bbufStride)     = b0;
#endif
  };

  stage(0, 0);
  int p = 0;
  for (int k0 = 0; k0 < K; k0 += 32) {
    const int nxt = k0 + 32;
    if (nxt < K) stage(p ^ 1, nxt);
#if HAVE_ASYNC
    if (nxt < K) __builtin_amdgcn_s_wait_asynccnt(3);
    else         __builtin_amdgcn_s_wait_asynccnt(0);
#endif
    __syncthreads();

    bf16x16 a0f = fragA(As[p], wm, lane);
    bf16x16 a1f = fragA(As[p], wm + 16, lane);
    bf16x16 b0f = fragB(Bs[p], wn, lane);
    bf16x16 b1f = fragB(Bs[p], wn + 16, lane);
    acc00 = wmma_bf16(a0f, b0f, acc00);
    acc01 = wmma_bf16(a0f, b1f, acc01);
    acc10 = wmma_bf16(a1f, b0f, acc10);
    acc11 = wmma_bf16(a1f, b1f, acc11);

    __syncthreads();
    p ^= 1;
  }

  // Epilogue: C layout (ISA): elem e -> row = e + (lane>=16 ? 8 : 0), col = lane%16
  const int colIn = lane & 15;
  const int rAdd  = (lane < 16) ? 0 : 8;
#pragma unroll
  for (int e = 0; e < 8; ++e) {
    int rr = e + rAdd;
    int rows[2] = { m0 + wm + rr, m0 + wm + 16 + rr };
    int cols[2] = { n0 + wn + colIn, n0 + wn + 16 + colIn };
    float vals[2][2] = { { acc00[e], acc01[e] }, { acc10[e], acc11[e] } };
#pragma unroll
    for (int i = 0; i < 2; ++i) {
#pragma unroll
      for (int j = 0; j < 2; ++j) {
        int r = rows[i], c = cols[j];
        if (r < M && c < N) {
          float v = vals[i][j] + bias[c];
          if (ACT == 1) v = gelu_exact(v);
          C[(size_t)r * N + c] = v;
        }
      }
    }
  }
}

// ---------------------------------------------------------------------------
// Series decomposition: moving average (window 25, edge-replicated)
// ---------------------------------------------------------------------------
__global__ __launch_bounds__(256) void decomp_kernel(const float* __restrict__ x,
                                                     float* __restrict__ trend,
                                                     float* __restrict__ seasonal) {
  int gid = blockIdx.x * 256 + threadIdx.x;
  if (gid >= B_ * T_ * FIN_) return;
  int f = gid & 31, tt = (gid >> 5) & 511, b = gid >> 14;
  float s = 0.f;
  for (int j = -12; j <= 12; ++j) {
    int ti = tt + j;
    ti = ti < 0 ? 0 : (ti > T_ - 1 ? T_ - 1 : ti);
    s += x[((size_t)(b * T_ + ti)) * FIN_ + f];
  }
  float tr = s * (1.0f / 25.0f);
  trend[gid]    = tr;
  seasonal[gid] = x[gid] - tr;
}

// ---------------------------------------------------------------------------
// Token conv embedding (wrap pad per reference) + positional encoding
// ---------------------------------------------------------------------------
__global__ __launch_bounds__(256) void embed_kernel(const float* __restrict__ seasonal,
                                                    const float* __restrict__ tokw, // (D,FIN,3)
                                                    float* __restrict__ h) {
  int gid = blockIdx.x * 256 + threadIdx.x;
  if (gid >= B_ * T_ * D_) return;
  int o = gid & 511, tt = (gid >> 9) & 511, b = gid >> 18;
  float acc = 0.f;
#pragma unroll
  for (int k = 0; k < 3; ++k) {
    int ti = tt + k - 1;
    ti = (ti < 0) ? T_ - 1 : (ti > T_ - 1 ? 0 : ti);
    const float* sp = seasonal + ((size_t)(b * T_ + ti)) * FIN_;
    const float* wp = tokw + (size_t)o * (FIN_ * 3) + k;
    for (int f = 0; f < FIN_; ++f) acc += sp[f] * wp[f * 3];
  }
  float dv = expf((float)(o & ~1) * (-9.210340371976184f / (float)D_));
  float pe = (o & 1) ? cosf((float)tt * dv) : sinf((float)tt * dv);
  h[gid] = acc + pe;
}

// ---------------------------------------------------------------------------
// Sampled QK scoring: M = max_s(qk) - mean_s(qk)
// ---------------------------------------------------------------------------
__global__ __launch_bounds__(256) void qk_sample_kernel(const float* __restrict__ Q,
                                                        const float* __restrict__ K,
                                                        const int* __restrict__ sidx, // (T, NTOP)
                                                        float* __restrict__ Mout) {
  int gid = blockIdx.x * 256 + threadIdx.x;
  if (gid >= B_ * H_ * T_) return;
  int l = gid & 511, hh = (gid >> 9) & 7, b = gid >> 12;
  const float* qp = Q + ((size_t)(b * T_ + l)) * D_ + hh * DH_;
  float mx = -3.4e38f, sm = 0.f;
  for (int s = 0; s < NTOP_; ++s) {
    int j = sidx[l * NTOP_ + s];
    const float* kp = K + ((size_t)(b * T_ + j)) * D_ + hh * DH_;
    float d = 0.f;
    for (int e = 0; e < DH_; ++e) d += qp[e] * kp[e];
    mx = fmaxf(mx, d);
    sm += d;
  }
  Mout[gid] = mx - sm * (1.0f / (float)NTOP_);
}

// ---------------------------------------------------------------------------
// Top-NTOP selection: one wave32 per (b,h)
// ---------------------------------------------------------------------------
__global__ __launch_bounds__(32) void topk_kernel(const float* __restrict__ Min,
                                                  int* __restrict__ topk) {
  __shared__ float s[T_];
  int bh = blockIdx.x;
  const float* m = Min + (size_t)bh * T_;
  for (int i = threadIdx.x; i < T_; i += 32) s[i] = m[i];
  __syncthreads();
  for (int it = 0; it < NTOP_; ++it) {
    float best = -3.4e38f; int bi = 0;
    for (int i = threadIdx.x; i < T_; i += 32) {
      float v = s[i];
      if (v > best || (v == best && i < bi)) { best = v; bi = i; }
    }
    for (int off = 16; off > 0; off >>= 1) {
      float ov = __shfl_xor(best, off, 32);
      int   oi = __shfl_xor(bi, off, 32);
      if (ov > best || (ov == best && oi < bi)) { best = ov; bi = oi; }
    }
    if (threadIdx.x == 0) topk[bh * NTOP_ + it] = bi;
    s[bi] = -3.4e38f;
    __syncthreads();
  }
}

// ---------------------------------------------------------------------------
// meanV and ctx fill
// ---------------------------------------------------------------------------
__global__ __launch_bounds__(256) void meanv_kernel(const float* __restrict__ V,
                                                    float* __restrict__ meanV) {
  int gid = blockIdx.x * 256 + threadIdx.x;
  if (gid >= B_ * D_) return;
  int c = gid & 511, b = gid >> 9;
  float s = 0.f;
  for (int l = 0; l < T_; ++l) s += V[((size_t)(b * T_ + l)) * D_ + c];
  meanV[gid] = s * (1.0f / (float)T_);
}

__global__ __launch_bounds__(256) void ctxfill_kernel(const float* __restrict__ meanV,
                                                      float* __restrict__ ctx) {
  int gid = blockIdx.x * 256 + threadIdx.x;
  if (gid >= B_ * T_ * D_) return;
  int c = gid & 511, row = gid >> 9, b = row >> 9;
  ctx[gid] = meanV[b * D_ + c];
}

// ---------------------------------------------------------------------------
// Attention update for selected queries: softmax(q_red K^T / 8) V, scattered
// ---------------------------------------------------------------------------
__global__ __launch_bounds__(256) void attn_upd_kernel(const float* __restrict__ Q,
                                                       const float* __restrict__ K,
                                                       const float* __restrict__ V,
                                                       const int* __restrict__ topk,
                                                       float* __restrict__ ctx) {
  __shared__ float qs[DH_];
  __shared__ float sc[T_];
  __shared__ float red[8];
  int u = blockIdx.x, hh = blockIdx.y, b = blockIdx.z;
  int t = threadIdx.x;
  int l_u = topk[(b * H_ + hh) * NTOP_ + u];
  if (t < DH_) qs[t] = Q[((size_t)(b * T_ + l_u)) * D_ + hh * DH_ + t];
  __syncthreads();
  for (int l = t; l < T_; l += 256) {
    const float* kp = K + ((size_t)(b * T_ + l)) * D_ + hh * DH_;
    float d = 0.f;
    for (int e = 0; e < DH_; ++e) d += qs[e] * kp[e];
    sc[l] = d * 0.125f;
  }
  __syncthreads();
  float lm = fmaxf(sc[t], sc[t + 256]);
  for (int off = 16; off > 0; off >>= 1) lm = fmaxf(lm, __shfl_xor(lm, off, 32));
  if ((t & 31) == 0) red[t >> 5] = lm;
  __syncthreads();
  float gmax = red[0];
  for (int w = 1; w < 8; ++w) gmax = fmaxf(gmax, red[w]);
  float e0 = expf(sc[t] - gmax), e1 = expf(sc[t + 256] - gmax);
  __syncthreads();
  sc[t] = e0; sc[t + 256] = e1;
  float ls = e0 + e1;
  for (int off = 16; off > 0; off >>= 1) ls += __shfl_xor(ls, off, 32);
  if ((t & 31) == 0) red[t >> 5] = ls;
  __syncthreads();
  float gsum = 0.f;
  for (int w = 0; w < 8; ++w) gsum += red[w];
  float inv = 1.0f / gsum;
  __syncthreads();
  if (t < DH_) {
    float acc = 0.f;
    for (int l = 0; l < T_; ++l) acc += sc[l] * V[((size_t)(b * T_ + l)) * D_ + hh * DH_ + t];
    ctx[((size_t)(b * T_ + l_u)) * D_ + hh * DH_ + t] = acc * inv;
  }
}

// ---------------------------------------------------------------------------
// Fused (a [+ b]) -> LayerNorm -> [ReLU]; one wave32 per 512-wide row
// ---------------------------------------------------------------------------
__global__ __launch_bounds__(256) void ln_kernel(const float* __restrict__ a,
                                                 const float* __restrict__ b,
                                                 const float* __restrict__ g,
                                                 const float* __restrict__ be,
                                                 float* __restrict__ out,
                                                 int rows, int relu) {
  int lane = threadIdx.x & 31, wid = threadIdx.x >> 5;
  int row = blockIdx.x * 8 + wid;
  if (row >= rows) return;
  const float* ap = a + (size_t)row * D_;
  const float* bp = b ? b + (size_t)row * D_ : nullptr;
  float x[16], s = 0.f, s2 = 0.f;
#pragma unroll
  for (int i = 0; i < 16; ++i) {
    int c = lane + i * 32;
    float v = ap[c] + (bp ? bp[c] : 0.f);
    x[i] = v; s += v; s2 += v * v;
  }
  for (int off = 16; off > 0; off >>= 1) {
    s  += __shfl_xor(s, off, 32);
    s2 += __shfl_xor(s2, off, 32);
  }
  float mu  = s * (1.0f / (float)D_);
  float var = s2 * (1.0f / (float)D_) - mu * mu;
  float r   = rsqrtf(var + 1e-5f);
#pragma unroll
  for (int i = 0; i < 16; ++i) {
    int c = lane + i * 32;
    float v = (x[i] - mu) * r * g[c] + be[c];
    if (relu) v = fmaxf(v, 0.f);
    out[(size_t)row * D_ + c] = v;
  }
}

__global__ __launch_bounds__(256) void add_kernel(const float* __restrict__ a,
                                                  const float* __restrict__ b,
                                                  float* __restrict__ out, int n) {
  int i = blockIdx.x * 256 + threadIdx.x;
  if (i < n) out[i] = a[i] + b[i];
}

__global__ __launch_bounds__(256) void glu_kernel(const float* __restrict__ g,
                                                  float* __restrict__ u, int rows) {
  int gid = blockIdx.x * 256 + threadIdx.x;
  if (gid >= rows * D_) return;
  int c = gid & 511, r = gid >> 9;
  float av = g[(size_t)r * (2 * D_) + c];
  float gv = g[(size_t)r * (2 * D_) + D_ + c];
  u[gid] = av * (1.0f / (1.0f + expf(-gv)));
}

__global__ __launch_bounds__(256) void dwconv_kernel(const float* __restrict__ u,
                                                     const float* __restrict__ dw, // (D,3)
                                                     const float* __restrict__ db,
                                                     float* __restrict__ out) {
  int gid = blockIdx.x * 256 + threadIdx.x;
  if (gid >= B_ * T_ * D_) return;
  int c = gid & 511, row = gid >> 9, l = row & 511;
  float left  = (l > 0)      ? u[(size_t)(row - 1) * D_ + c] : 0.f;
  float mid   = u[(size_t)row * D_ + c];
  float right = (l < T_ - 1) ? u[(size_t)(row + 1) * D_ + c] : 0.f;
  out[gid] = left * dw[c * 3 + 0] + mid * dw[c * 3 + 1] + right * dw[c * 3 + 2] + db[c];
}

// (B,T,D) -> (B,D,T) tiled transpose
__global__ __launch_bounds__(256) void transpose_kernel(const float* __restrict__ in,
                                                        float* __restrict__ out) {
  __shared__ float tile[32][33];
  int b = blockIdx.z;
  int t0 = blockIdx.x * 32, d0 = blockIdx.y * 32;
  int lx = threadIdx.x & 31, ly = threadIdx.x >> 5;
#pragma unroll
  for (int i = 0; i < 32; i += 8)
    tile[ly + i][lx] = in[(size_t)b * (T_ * D_) + (size_t)(t0 + ly + i) * D_ + d0 + lx];
  __syncthreads();
#pragma unroll
  for (int i = 0; i < 32; i += 8)
    out[(size_t)b * (T_ * D_) + (size_t)(d0 + ly + i) * T_ + t0 + lx] = tile[lx][ly + i];
}

// ---------------------------------------------------------------------------
// Final head: out[b,p] = dec[b,p,:]·fc_w[0,:] + fc_b[0] + w_dec[0]*trend_out[b,p,0]
// ---------------------------------------------------------------------------
__global__ __launch_bounds__(256) void final_kernel(const float* __restrict__ dec_t,
                                                    const float* __restrict__ fcw,
                                                    const float* __restrict__ fcb,
                                                    const float* __restrict__ wdec,
                                                    const float* __restrict__ trend,
                                                    const float* __restrict__ trw,
                                                    const float* __restrict__ trb,
                                                    float* __restrict__ out) {
  int gid = blockIdx.x * 256 + threadIdx.x;
  if (gid >= B_ * PRED_) return;
  int p = gid % PRED_, b = gid / PRED_;
  float acc = 0.f;
  for (int d = 0; d < D_; ++d)
    acc += dec_t[((size_t)(b * D_ + d)) * PRED_ + p] * fcw[d];
  acc += fcb[0];
  float tr = 0.f;
  for (int tt = 0; tt < T_; ++tt)
    tr += trend[((size_t)(b * T_ + tt)) * FIN_ + 0] * trw[p * T_ + tt];
  tr += trb[p];
  out[gid] = acc + wdec[0] * tr;
}

// ---------------------------------------------------------------------------
// Host launcher
// ---------------------------------------------------------------------------
extern "C" void kernel_launch(void* const* d_in, const int* in_sizes, int n_in,
                              void* d_out, int out_size, void* d_ws, size_t ws_size,
                              hipStream_t stream) {
  (void)in_sizes; (void)n_in; (void)out_size; (void)ws_size;
  const float* x          = (const float*)d_in[0];
  const int*   sample_idx = (const int*)d_in[2];
  auto F = [&](int i) { return (const float*)d_in[i]; };
  const float* token_w = F(3);
  const float* trend_w = F(4);
  const float* trend_b = F(5);
  const float* enc_g   = F(6);
  const float* enc_be  = F(7);
  const float* mlp_w1  = F(8);
  const float* mlp_b1  = F(9);
  const float* mlp_w2  = F(10);
  const float* mlp_b2  = F(11);
  const float* fc_w    = F(12);
  const float* fc_b    = F(13);
  const float* w_dec   = F(14);
  auto LW = [&](int layer, int j) { return (const float*)d_in[15 + layer * 26 + j]; };

  char* ws = (char*)d_ws;
  const size_t SZ_BTD = (size_t)B_ * T_ * D_ * 4;  // 16 MB
  float* h      = (float*)(ws);
  float* bufB   = (float*)(ws + 1 * SZ_BTD);
  float* bufC   = (float*)(ws + 2 * SZ_BTD);
  float* bufQ   = (float*)(ws + 3 * SZ_BTD);
  float* bufK   = (float*)(ws + 4 * SZ_BTD);
  float* bufV   = (float*)(ws + 5 * SZ_BTD);
  float* bufBig = (float*)(ws + 6 * SZ_BTD);       // 64 MB (B*T*2048 f32)
  size_t off    = 10 * SZ_BTD;
  float* seasonal = (float*)(ws + off); off += (size_t)B_ * T_ * FIN_ * 4;
  float* trend    = (float*)(ws + off); off += (size_t)B_ * T_ * FIN_ * 4;
  float* Mbuf     = (float*)(ws + off); off += (size_t)B_ * H_ * T_ * 4;
  int*   topk     = (int*)  (ws + off); off += 32768;
  float* meanV    = (float*)(ws + off); off += 32768;
  u16*   Abf      = (u16*)  (ws + off); off += (size_t)8192 * 2048 * 2;  // 32 MB bf16 act
  u16*   wbf      = (u16*)  (ws + off);

  auto cvt = [&](const float* src, size_t& wofs, int n) -> u16* {
    u16* p = wbf + wofs; wofs += (size_t)n;
    convert_bf16_kernel<<<dim3((n + 255) / 256), dim3(256), 0, stream>>>(src, p, n);
    return p;
  };
  size_t wofs = 0;
  u16 *Wq[2], *Wk[2], *Wv[2], *Wo[2], *Wp1[2], *Wp2[2], *Wf1[2], *Wf2[2];
  for (int l = 0; l < 2; ++l) {
    Wq[l]  = cvt(LW(l, 0),  wofs, D_ * D_);
    Wk[l]  = cvt(LW(l, 2),  wofs, D_ * D_);
    Wv[l]  = cvt(LW(l, 4),  wofs, D_ * D_);
    Wo[l]  = cvt(LW(l, 6),  wofs, D_ * D_);
    Wp1[l] = cvt(LW(l, 16), wofs, 2 * D_ * D_);
    Wp2[l] = cvt(LW(l, 20), wofs, D_ * D_);
    Wf1[l] = cvt(LW(l, 22), wofs, DFF_ * D_);
    Wf2[l] = cvt(LW(l, 24), wofs, D_ * DFF_);
  }
  u16* Wm1 = cvt(mlp_w1, wofs, D_ * T_);
  u16* Wm2 = cvt(mlp_w2, wofs, PRED_ * D_);

  // Convert one GEMM A-operand into the shared bf16 staging buffer
  auto cvtA = [&](const float* src, int n) {
    convert_bf16_kernel<<<dim3((n + 255) / 256), dim3(256), 0, stream>>>(src, Abf, n);
  };
  auto gemm = [&](const u16* W, const float* bias, float* C, int M, int N, int K, int act) {
    dim3 grid((N + 63) / 64, (M + 127) / 128);
    if (act)
      gemm_wmma<1><<<grid, 256, 0, stream>>>(Abf, W, bias, C, M, N, K);
    else
      gemm_wmma<0><<<grid, 256, 0, stream>>>(Abf, W, bias, C, M, N, K);
  };

  const int M = B_ * T_;                 // 8192 (== B_*D_ for the head)
  const int NBTD = B_ * T_ * D_;         // 4,194,304
  const dim3 gBTD((NBTD + 255) / 256), b256(256);

  // 1) decomposition + embedding
  decomp_kernel<<<dim3((B_ * T_ * FIN_ + 255) / 256), b256, 0, stream>>>(x, trend, seasonal);
  embed_kernel<<<gBTD, b256, 0, stream>>>(seasonal, token_w, h);

  // 2) encoder layers
  for (int l = 0; l < 2; ++l) {
    // attention branch
    cvtA(h, NBTD);
    gemm(Wq[l], LW(l, 1), bufQ, M, D_, D_, 0);
    gemm(Wk[l], LW(l, 3), bufK, M, D_, D_, 0);
    gemm(Wv[l], LW(l, 5), bufV, M, D_, D_, 0);
    qk_sample_kernel<<<dim3(B_ * H_ * T_ / 256), b256, 0, stream>>>(
        bufQ, bufK, sample_idx + (size_t)l * T_ * NTOP_, Mbuf);
    topk_kernel<<<dim3(B_ * H_), dim3(32), 0, stream>>>(Mbuf, topk);
    meanv_kernel<<<dim3(B_ * D_ / 256), b256, 0, stream>>>(bufV, meanV);
    ctxfill_kernel<<<gBTD, b256, 0, stream>>>(meanV, bufC);
    attn_upd_kernel<<<dim3(NTOP_, H_, B_), b256, 0, stream>>>(bufQ, bufK, bufV, topk, bufC);
    cvtA(bufC, NBTD);
    gemm(Wo[l], LW(l, 7), bufB, M, D_, D_, 0);                        // new_h
    ln_kernel<<<dim3(M / 8), b256, 0, stream>>>(h, bufB, LW(l, 8), LW(l, 9), bufB, M, 0);

    // conv branch
    cvtA(h, NBTD);
    gemm(Wp1[l], LW(l, 17), bufBig, M, 2 * D_, D_, 0);
    glu_kernel<<<gBTD, b256, 0, stream>>>(bufBig, bufQ, M);
    dwconv_kernel<<<gBTD, b256, 0, stream>>>(bufQ, LW(l, 18), LW(l, 19), bufK);
    ln_kernel<<<dim3(M / 8), b256, 0, stream>>>(bufK, nullptr, LW(l, 14), LW(l, 15), bufK, M, 1);
    cvtA(bufK, NBTD);
    gemm(Wp2[l], LW(l, 21), bufV, M, D_, D_, 0);                      // u
    add_kernel<<<gBTD, b256, 0, stream>>>(bufV, h, bufV, NBTD);        // conv_x = h + u
    ln_kernel<<<dim3(M / 8), b256, 0, stream>>>(bufV, h, LW(l, 10), LW(l, 11), bufV, M, 0);

    // combine + FFN
    add_kernel<<<gBTD, b256, 0, stream>>>(bufB, bufV, bufC, NBTD);     // output
    cvtA(bufC, NBTD);
    gemm(Wf1[l], LW(l, 23), bufBig, M, DFF_, D_, 1);                   // gelu(ff1)
    cvtA(bufBig, M * DFF_);
    gemm(Wf2[l], LW(l, 25), bufQ, M, D_, DFF_, 0);                     // ff2
    ln_kernel<<<dim3(M / 8), b256, 0, stream>>>(bufC, bufQ, LW(l, 12), LW(l, 13), h, M, 0);
  }

  // 3) head
  ln_kernel<<<dim3(M / 8), b256, 0, stream>>>(h, nullptr, enc_g, enc_be, bufB, M, 0);
  transpose_kernel<<<dim3(16, 16, B_), b256, 0, stream>>>(bufB, bufC);   // (B,D,T)
  cvtA(bufC, NBTD);
  gemm(Wm1, mlp_b1, bufQ, B_ * D_, D_, T_, 1);                          // gelu(mlp1)
  cvtA(bufQ, NBTD);
  gemm(Wm2, mlp_b2, bufV, B_ * D_, PRED_, D_, 0);                       // dec_t (B*D, 96)
  final_kernel<<<dim3((B_ * PRED_ + 255) / 256), b256, 0, stream>>>(
      bufV, fc_w, fc_b, w_dec, trend, trend_w, trend_b, (float*)d_out);
}
